// TinyTribeMoE_77214922047958
// MI455X (gfx1250) — compile-verified, matrix operand order checked by reading, not run on hardware
//
#include <hip/hip_runtime.h>
#include <math.h>

// ---------------------------------------------------------------------------
// TinyTribeMoE forward for MI455X (gfx1250, wave32, WMMA).
// All dense GEMMs go through v_wmma_f32_16x16x32_f16 (f16 A/B, f32 accum).
// ---------------------------------------------------------------------------

typedef __attribute__((ext_vector_type(16))) _Float16 v16h;
typedef __attribute__((ext_vector_type(4)))  _Float16 v4h;
typedef __attribute__((ext_vector_type(2)))  __fp16   v2fp;
typedef __attribute__((ext_vector_type(8)))  float    v8f;

// Model dims
#define BB 2
#define TT 128
#define SS 384           // T*3
#define DD 512
#define HH 8
#define DH 64
#define EE 8
#define FF_ 1024
#define PI_ 768
#define LR_ 256
#define NV_ 5124
#define NTOK (BB*SS)     // 768 rows in the trunk
#define NPTOK (BB*TT)    // 256 rows per projector

// GEMM tiling
#define TM 64
#define TN 128
#define TK 32
#define APAD 16          // As row stride = 48 halves = 96B (32B aligned)
#define BPAD 8

enum { EPI_BIAS = 0, EPI_GELU = 1, EPI_RES = 2, EPI_MOE = 3 };

__device__ __forceinline__ float ttm_gelu(float x) {
  return 0.5f * x * (1.0f + erff(x * 0.70710678118654752f));
}

__device__ __forceinline__ v8f ttm_wmma(v16h a, v16h b, v8f c) {
  // v_wmma_f32_16x16x32_f16  (neg_a, A, neg_b, B, c_mod, C, reuse_a, reuse_b)
  return __builtin_amdgcn_wmma_f32_16x16x32_f16(false, a, false, b, (short)0, c,
                                                false, false);
}

// pack 4 f32 -> 4 f16 (v_cvt_pk_rtz_f16_f32 x2)
__device__ __forceinline__ v4h ttm_pk4(float x, float y, float z, float w) {
  union { v4h h4; v2fp h2[2]; } u;
  u.h2[0] = __builtin_amdgcn_cvt_pkrtz(x, y);
  u.h2[1] = __builtin_amdgcn_cvt_pkrtz(z, w);
  return u.h4;
}

// ---------------------------------------------------------------------------
// Generic WMMA GEMM: C[M,N] (+epilogue) = A[M,K] (f32,row-major) * B[K,N]
// A/B staged through LDS as f16. M must be a multiple of 64, K of 32, and
// every lda/ldb/N a multiple of 4 (true for all call sites; N=5124 edge tile
// handled since quads never straddle N). A is stored in LDS with the WMMA
// fragment K-permutation (8-groups reordered {0,2,1,3}) so each lane's 16
// fragment halves are contiguous -> 2x ds_load_b128 per A fragment.
// Full tiles (n0+TN <= N) take a block-uniform guard-free path: no divergent
// exec masking in the K loop or epilogue for any GEMM except the single
// ragged NV=5124 edge tile.
// gather!=nullptr: B/bias are indexed by gather[blockIdx.z] (subject weights).
// TRANS: store C at [n*ldc + m] (prediction transpose).
// ---------------------------------------------------------------------------
template <int EPI, bool TRANS>
__global__ __launch_bounds__(256) void ttm_gemm_kernel(
    const float* __restrict__ A, int lda, long sA,
    const float* __restrict__ B, int ldb, long sB,
    const float* __restrict__ bias, long sBias,
    const float* __restrict__ rowscale, int rsStride,
    float* __restrict__ C, int ldc, long sC,
    int M, int N, int K, const int* __restrict__ gather) {
  __shared__ __align__(32) _Float16 As[TM][TK + APAD];
  __shared__ __align__(16) _Float16 Bs[TK][TN + BPAD];

  const int bz = blockIdx.z;
  const int gi = gather ? gather[bz] : bz;
  A += (long)bz * sA;
  B += (long)gi * sB;
  const float* biasp = bias ? (bias + (long)gi * sBias) : nullptr;
  C += (long)bz * sC;

  const int m0 = blockIdx.y * TM;
  const int n0 = blockIdx.x * TN;
  const int tid = threadIdx.x;
  const int lane = tid & 31, wave = tid >> 5;
  const int wm = wave >> 2, wn = wave & 3;  // 2x4 wave grid -> 32x32 per wave
  const int r16 = lane & 15, hi = lane >> 4;
  const bool full = (n0 + TN) <= N;  // block-uniform

  v8f acc[2][2] = {};

  for (int k0 = 0; k0 < K; k0 += TK) {
    // A tile: 64x32 f32 -> LDS f16 (perm-K). 512 float4 quads, 2 per thread.
#pragma unroll
    for (int i = 0; i < 2; ++i) {
      int idx = i * 256 + tid;
      int r = idx >> 3;  // 8 quads per row
      int q = idx & 7;
      const float4 f =
          *(const float4*)(A + (long)(m0 + r) * lda + (k0 + q * 4));
      int g = q >> 1;                       // 8-group index 0..3
      int gp = ((g & 1) << 1) | (g >> 1);   // fragment group order {0,2,1,3}
      int kpos = gp * 8 + (q & 1) * 4;
      *(v4h*)&As[r][kpos] = ttm_pk4(f.x, f.y, f.z, f.w);
    }
    // B tile: 32x128 f32 -> LDS f16. 1024 quads, 4 per thread.
    if (full) {
#pragma unroll
      for (int i = 0; i < 4; ++i) {
        int idx = i * 256 + tid;
        int r = idx >> 5;  // 32 quads per row
        int q = idx & 31;
        const float4 f =
            *(const float4*)(B + (long)(k0 + r) * ldb + (n0 + q * 4));
        *(v4h*)&Bs[r][q * 4] = ttm_pk4(f.x, f.y, f.z, f.w);
      }
    } else {
#pragma unroll
      for (int i = 0; i < 4; ++i) {
        int idx = i * 256 + tid;
        int r = idx >> 5;
        int q = idx & 31;
        int n = n0 + q * 4;
        float4 f;
        if (n < N) {
          f = *(const float4*)(B + (long)(k0 + r) * ldb + n);
        } else {
          f.x = 0.0f; f.y = 0.0f; f.z = 0.0f; f.w = 0.0f;
        }
        *(v4h*)&Bs[r][q * 4] = ttm_pk4(f.x, f.y, f.z, f.w);
      }
    }
    __syncthreads();

    v16h a[2], b[2];
#pragma unroll
    for (int mi = 0; mi < 2; ++mi) {
      int row = wm * 32 + mi * 16 + r16;
      a[mi] = *(const v16h*)&As[row][hi * 16];  // contiguous: 2x ds_load_b128
    }
#pragma unroll
    for (int ni = 0; ni < 2; ++ni) {
      int col = wn * 32 + ni * 16 + r16;
#pragma unroll
      for (int e = 0; e < 16; ++e) {
        int kk = hi * 8 + (e < 8 ? e : e + 8);  // ISA 16-bit 16x32 lane map
        b[ni][e] = Bs[kk][col];
      }
    }
#pragma unroll
    for (int mi = 0; mi < 2; ++mi)
#pragma unroll
      for (int ni = 0; ni < 2; ++ni)
        acc[mi][ni] = ttm_wmma(a[mi], b[ni], acc[mi][ni]);
    __syncthreads();
  }

  // C/D layout: VGPR r, lanes 0-15 -> M = r, N = lane; lanes 16-31 -> M = 8+r
#pragma unroll
  for (int mi = 0; mi < 2; ++mi) {
#pragma unroll
    for (int ni = 0; ni < 2; ++ni) {
#pragma unroll
      for (int r = 0; r < 8; ++r) {
        int m = m0 + wm * 32 + mi * 16 + hi * 8 + r;
        int n = n0 + wn * 32 + ni * 16 + r16;
        if (full || n < N) {  // M always a multiple of TM at our call sites
          float v = acc[mi][ni][r];
          if (biasp) v += biasp[n];
          long off = TRANS ? ((long)n * ldc + m) : ((long)m * ldc + n);
          if (EPI == EPI_GELU) v = ttm_gelu(v);
          if (EPI == EPI_RES) v += C[off];
          if (EPI == EPI_MOE) v = C[off] + rowscale[(long)m * rsStride] * v;
          C[off] = v;
        }
      }
    }
  }
}

// ---------------------------------------------------------------------------
// LayerNorm: one 256-thread block per row, population variance, eps=1e-5
// ---------------------------------------------------------------------------
__global__ __launch_bounds__(256) void ttm_layernorm_kernel(
    const float* __restrict__ x, const float* __restrict__ s,
    const float* __restrict__ b, float* __restrict__ y, int cols) {
  const int row = blockIdx.x;
  const float* xr = x + (long)row * cols;
  float* yr = y + (long)row * cols;
  __shared__ float red[8];
  const int tid = threadIdx.x, lane = tid & 31, wave = tid >> 5;

  float sum = 0.0f;
  for (int c = tid; c < cols; c += 256) sum += xr[c];
  for (int o = 16; o > 0; o >>= 1) sum += __shfl_xor(sum, o, 32);
  if (lane == 0) red[wave] = sum;
  __syncthreads();
  if (tid == 0) {
    float t = 0;
    for (int i = 0; i < 8; ++i) t += red[i];
    red[0] = t;
  }
  __syncthreads();
  const float mean = red[0] / (float)cols;
  __syncthreads();

  float vs = 0.0f;
  for (int c = tid; c < cols; c += 256) {
    float d = xr[c] - mean;
    vs += d * d;
  }
  for (int o = 16; o > 0; o >>= 1) vs += __shfl_xor(vs, o, 32);
  if (lane == 0) red[wave] = vs;
  __syncthreads();
  if (tid == 0) {
    float t = 0;
    for (int i = 0; i < 8; ++i) t += red[i];
    red[0] = t;
  }
  __syncthreads();
  const float rstd = rsqrtf(red[0] / (float)cols + 1e-5f);
  for (int c = tid; c < cols; c += 256)
    yr[c] = (xr[c] - mean) * rstd * s[c] + b[c];
}

// ---------------------------------------------------------------------------
// Fused naive MHA: grid.x = B*H, 8 waves, each wave owns one q row at a time.
// q/k/v/o are [B, S, D] with head slice at h*64. S=384, dh=64, scale=1/8.
// ---------------------------------------------------------------------------
__global__ __launch_bounds__(256) void ttm_attention_kernel(
    const float* __restrict__ q, const float* __restrict__ k,
    const float* __restrict__ v, float* __restrict__ o) {
  const int bh = blockIdx.x;
  const int b = bh >> 3, h = bh & 7;
  const float* qb = q + (long)b * SS * DD + h * DH;
  const float* kb = k + (long)b * SS * DD + h * DH;
  const float* vb = v + (long)b * SS * DD + h * DH;
  float* ob = o + (long)b * SS * DD + h * DH;

  __shared__ float sc[8][SS];
  __shared__ float qs[8][DH];
  const int tid = threadIdx.x, lane = tid & 31, wave = tid >> 5;
  const float scale = 0.125f;  // 1/sqrt(64)

  for (int qi = wave; qi < SS; qi += 8) {  // 48 uniform iterations
    qs[wave][lane * 2 + 0] = qb[(long)qi * DD + lane * 2 + 0];
    qs[wave][lane * 2 + 1] = qb[(long)qi * DD + lane * 2 + 1];
    __syncthreads();
    for (int kk = lane; kk < SS; kk += 32) {
      float dot = 0.0f;
#pragma unroll 8
      for (int d = 0; d < DH; ++d) dot += qs[wave][d] * kb[(long)kk * DD + d];
      sc[wave][kk] = dot * scale;
    }
    __syncthreads();
    float mx = -1e30f;
    for (int kk = lane; kk < SS; kk += 32) mx = fmaxf(mx, sc[wave][kk]);
    for (int off = 16; off > 0; off >>= 1)
      mx = fmaxf(mx, __shfl_xor(mx, off, 32));
    float sum = 0.0f;
    for (int kk = lane; kk < SS; kk += 32) {
      float e = expf(sc[wave][kk] - mx);
      sc[wave][kk] = e;
      sum += e;
    }
    for (int off = 16; off > 0; off >>= 1) sum += __shfl_xor(sum, off, 32);
    __syncthreads();
    const float inv = 1.0f / sum;
    const int d0 = lane * 2, d1 = lane * 2 + 1;
    float a0 = 0.0f, a1 = 0.0f;
    for (int kk = 0; kk < SS; ++kk) {
      float p = sc[wave][kk];
      a0 += p * vb[(long)kk * DD + d0];
      a1 += p * vb[(long)kk * DD + d1];
    }
    ob[(long)qi * DD + d0] = a0 * inv;
    ob[(long)qi * DD + d1] = a1 * inv;
    __syncthreads();
  }
}

// Router gate logits: logits[row,e] = x[row,:] . gate[:,e]   (N=8 too small
// for WMMA tiles; plain dot kernel)
__global__ void ttm_gate_kernel(const float* __restrict__ x,
                                const float* __restrict__ g,
                                float* __restrict__ logits, int rows) {
  int idx = blockIdx.x * blockDim.x + threadIdx.x;
  if (idx >= rows * EE) return;
  int row = idx >> 3, e = idx & 7;
  float acc = 0.0f;
  for (int d = 0; d < DD; ++d) acc += x[(long)row * DD + d] * g[(long)d * EE + e];
  logits[idx] = acc;
}

// Top-2 router: combine weights + aux-loss statistics (atomics into aux[]):
// aux[0..7]=argmax counts, aux[8..15]=sum of softmax probs, aux[16]=sum l^2,
// aux[17]=running aux total across layers.
__global__ void ttm_router_kernel(const float* __restrict__ logits,
                                  float* __restrict__ combine,
                                  float* __restrict__ aux, int rows) {
  int row = blockIdx.x * blockDim.x + threadIdx.x;
  if (row >= rows) return;
  float l[EE];
#pragma unroll
  for (int e = 0; e < EE; ++e) l[e] = logits[(long)row * EE + e];
  int i1 = 0;
#pragma unroll
  for (int e = 1; e < EE; ++e)
    if (l[e] > l[i1]) i1 = e;
  int i2 = -1;
#pragma unroll
  for (int e = 0; e < EE; ++e) {
    if (e == i1) continue;
    if (i2 < 0 || l[e] > l[i2]) i2 = e;
  }
  float w2 = expf(l[i2] - l[i1]);
  float inv = 1.0f / (1.0f + w2);
#pragma unroll
  for (int e = 0; e < EE; ++e) combine[(long)row * EE + e] = 0.0f;
  combine[(long)row * EE + i1] = inv;
  combine[(long)row * EE + i2] = w2 * inv;

  float m = l[i1], s = 0.0f, p[EE];
#pragma unroll
  for (int e = 0; e < EE; ++e) {
    p[e] = expf(l[e] - m);
    s += p[e];
  }
  float sq = 0.0f;
#pragma unroll
  for (int e = 0; e < EE; ++e) {
    atomicAdd(&aux[8 + e], p[e] / s);
    sq += l[e] * l[e];
  }
  atomicAdd(&aux[i1], 1.0f);
  atomicAdd(&aux[16], sq);
}

__global__ void ttm_aux_layer_finalize(float* aux, float ntok) {
  if (blockIdx.x == 0 && threadIdx.x == 0) {
    float s = 0.0f;
    for (int e = 0; e < EE; ++e) s += (aux[e] / ntok) * (aux[8 + e] / ntok);
    aux[17] += (float)EE * s + (aux[16] / (ntok * (float)EE)) * 1e-3f;
  }
}

__global__ void ttm_aux_write(const float* aux, float* out) {
  if (blockIdx.x == 0 && threadIdx.x == 0) out[0] = aux[17] * 0.01f;
}

__global__ void ttm_zero_kernel(float* p, int n) {
  int i = blockIdx.x * blockDim.x + threadIdx.x;
  if (i < n) p[i] = 0.0f;
}

// fused[b, t*3+m, :] = proj[b*T+t, :] + mod_embed[m] + pos[t*3+m]
__global__ void ttm_assemble_kernel(const float* __restrict__ proj,
                                    const float* __restrict__ mode,
                                    const float* __restrict__ pos,
                                    float* __restrict__ fused, int modality) {
  int i = blockIdx.x * blockDim.x + threadIdx.x;
  if (i >= NPTOK * DD) return;
  int d = i & (DD - 1);
  int bt = i >> 9;
  int t = bt & (TT - 1), b = bt >> 7;
  int s = t * 3 + modality;
  fused[((long)(b * SS + s)) * DD + d] = proj[i] + mode[d] + pos[(long)s * DD + d];
}

// pooled[b,t,d] = mean over 3 modalities of final-LN trunk
__global__ void ttm_pool_kernel(const float* __restrict__ xln,
                                float* __restrict__ pooled) {
  int i = blockIdx.x * blockDim.x + threadIdx.x;
  if (i >= NPTOK * DD) return;
  int d = i & (DD - 1);
  int bt = i >> 9;
  int t = bt & (TT - 1), b = bt >> 7;
  const float* base = xln + ((long)(b * SS + t * 3)) * DD + d;
  pooled[i] = (base[0] + base[DD] + base[2 * DD]) * (1.0f / 3.0f);
}

// ---------------------------------------------------------------------------
// Host-side GEMM dispatch
// ---------------------------------------------------------------------------
static void ttm_gemm(hipStream_t st, int epi, bool trans, const float* A,
                     int lda, long sA, const float* B, int ldb, long sB,
                     const float* bias, long sBias, const float* rs,
                     int rsStride, float* C, int ldc, long sC, int M, int N,
                     int K, const int* gather, int batches) {
  dim3 grid((N + TN - 1) / TN, (M + TM - 1) / TM, batches), block(256);
  if (trans) {
    ttm_gemm_kernel<EPI_BIAS, true><<<grid, block, 0, st>>>(
        A, lda, sA, B, ldb, sB, bias, sBias, rs, rsStride, C, ldc, sC, M, N, K,
        gather);
    return;
  }
  switch (epi) {
    case EPI_BIAS:
      ttm_gemm_kernel<EPI_BIAS, false><<<grid, block, 0, st>>>(
          A, lda, sA, B, ldb, sB, bias, sBias, rs, rsStride, C, ldc, sC, M, N,
          K, gather);
      break;
    case EPI_GELU:
      ttm_gemm_kernel<EPI_GELU, false><<<grid, block, 0, st>>>(
          A, lda, sA, B, ldb, sB, bias, sBias, rs, rsStride, C, ldc, sC, M, N,
          K, gather);
      break;
    case EPI_RES:
      ttm_gemm_kernel<EPI_RES, false><<<grid, block, 0, st>>>(
          A, lda, sA, B, ldb, sB, bias, sBias, rs, rsStride, C, ldc, sC, M, N,
          K, gather);
      break;
    case EPI_MOE:
      ttm_gemm_kernel<EPI_MOE, false><<<grid, block, 0, st>>>(
          A, lda, sA, B, ldb, sB, bias, sBias, rs, rsStride, C, ldc, sC, M, N,
          K, gather);
      break;
  }
}

// ---------------------------------------------------------------------------
// Input index map. Assumption: the harness flattens the inputs as the four
// top-level tensors in setup_inputs() order, then `params` as a JAX pytree
// (dict keys sorted alphabetically at every level; lists in index order).
//   params sorted keys: audio, fn_b, fn_s, layers, lr_w, mod_embed, pos,
//                       subj_b, subj_w, text, video
//   projector sorted:   b1,b2,b3,ln1_b,ln1_s,ln2_b,ln2_s,w1,w2,w3      (10)
//   layer sorted: bk,bo,bq,bv,eb1,eb2,ew1,ew2,gate,n1_b,n1_s,n2_b,n2_s,
//                 wk,wo,wq,wv                                           (17)
// ---------------------------------------------------------------------------
#define IDX_TEXT_FEAT 0
#define IDX_AUDIO_FEAT 1
#define IDX_VIDEO_FEAT 2
#define IDX_SUBJECT 3
#define IDX_AUDIO_PROJ 4   // ..13
#define IDX_FN_B 14
#define IDX_FN_S 15
#define IDX_LAYER0 16      // 4 layers x 17 leaves -> ..83
#define IDX_LR_W 84
#define IDX_MOD_EMBED 85
#define IDX_POS 86
#define IDX_SUBJ_B 87
#define IDX_SUBJ_W 88
#define IDX_TEXT_PROJ 89   // ..98
#define IDX_VIDEO_PROJ 99  // ..108
// projector leaf offsets
#define P_B1 0
#define P_B2 1
#define P_B3 2
#define P_LN1B 3
#define P_LN1S 4
#define P_LN2B 5
#define P_LN2S 6
#define P_W1 7
#define P_W2 8
#define P_W3 9
// layer leaf offsets
#define L_BK 0
#define L_BO 1
#define L_BQ 2
#define L_BV 3
#define L_EB1 4
#define L_EB2 5
#define L_EW1 6
#define L_EW2 7
#define L_GATE 8
#define L_N1B 9
#define L_N1S 10
#define L_N2B 11
#define L_N2S 12
#define L_WK 13
#define L_WO 14
#define L_WQ 15
#define L_WV 16

extern "C" void kernel_launch(void* const* d_in, const int* in_sizes, int n_in,
                              void* d_out, int out_size, void* d_ws,
                              size_t ws_size, hipStream_t stream) {
  (void)in_sizes; (void)out_size; (void)ws_size;
  const float* F[128];
  for (int i = 0; i < n_in && i < 128; ++i) F[i] = (const float*)d_in[i];
  const int* subject_id = (const int*)d_in[IDX_SUBJECT];

  // Workspace layout (floats)
  float* ws = (float*)d_ws;
  size_t off = 0;
  float* bufA = ws + off;   off += (size_t)NTOK * FF_;   // 768x1024
  float* bufB = ws + off;   off += (size_t)NTOK * FF_;   // 768x1024
  float* fused = ws + off;  off += (size_t)NTOK * DD;    // 768x512
  float* xln = ws + off;    off += (size_t)NTOK * 640;   // LN scratch (max cols)
  float* qb = ws + off;     off += (size_t)NTOK * DD;
  float* kb = ws + off;     off += (size_t)NTOK * DD;
  float* vb = ws + off;     off += (size_t)NTOK * DD;
  float* ab = ws + off;     off += (size_t)NTOK * DD;
  float* logits = ws + off; off += (size_t)NTOK * EE;
  float* combine = ws + off; off += (size_t)NTOK * EE;
  float* bottl = ws + off;  off += (size_t)NPTOK * LR_;  // 256x256
  float* auxw = ws + off;   off += 32;

  // Output layout: prediction [2,5124,128] | pooled [2,128,512] | aux
  float* out_pred = (float*)d_out;
  float* out_pool = out_pred + (size_t)BB * NV_ * TT;
  float* out_aux = out_pool + (size_t)BB * TT * DD;

  ttm_zero_kernel<<<1, 32, 0, stream>>>(auxw, 32);

  // ---- Projectors ------------------------------------------------------
  const float* mod_embed = F[IDX_MOD_EMBED];
  const float* pos = F[IDX_POS];
  struct { int feat, base, in_dim, modality; } projs[3] = {
      {IDX_TEXT_FEAT, IDX_TEXT_PROJ, 384, 0},
      {IDX_AUDIO_FEAT, IDX_AUDIO_PROJ, 384, 1},
      {IDX_VIDEO_FEAT, IDX_VIDEO_PROJ, 640, 2}};
  for (int p = 0; p < 3; ++p) {
    int base = projs[p].base, in_dim = projs[p].in_dim;
    ttm_layernorm_kernel<<<NPTOK, 256, 0, stream>>>(
        F[projs[p].feat], F[base + P_LN1S], F[base + P_LN1B], xln, in_dim);
    ttm_gemm(stream, EPI_GELU, false, xln, in_dim, 0, F[base + P_W1], PI_, 0,
             F[base + P_B1], 0, nullptr, 0, bufA, PI_, 0, NPTOK, PI_, in_dim,
             nullptr, 1);
    ttm_gemm(stream, EPI_GELU, false, bufA, PI_, 0, F[base + P_W2], PI_, 0,
             F[base + P_B2], 0, nullptr, 0, bufB, PI_, 0, NPTOK, PI_, PI_,
             nullptr, 1);
    ttm_gemm(stream, EPI_BIAS, false, bufB, PI_, 0, F[base + P_W3], DD, 0,
             F[base + P_B3], 0, nullptr, 0, bufA, DD, 0, NPTOK, DD, PI_,
             nullptr, 1);
    ttm_layernorm_kernel<<<NPTOK, 256, 0, stream>>>(
        bufA, F[base + P_LN2S], F[base + P_LN2B], bufB, DD);
    ttm_assemble_kernel<<<(NPTOK * DD + 255) / 256, 256, 0, stream>>>(
        bufB, mod_embed + projs[p].modality * DD, pos, fused,
        projs[p].modality);
  }

  // ---- Transformer layers ---------------------------------------------
  for (int L = 0; L < 4; ++L) {
    int lb = IDX_LAYER0 + 17 * L;
    // pre-attn LN
    ttm_layernorm_kernel<<<NTOK, 256, 0, stream>>>(fused, F[lb + L_N1S],
                                                   F[lb + L_N1B], xln, DD);
    // QKV projections
    ttm_gemm(stream, EPI_BIAS, false, xln, DD, 0, F[lb + L_WQ], DD, 0,
             F[lb + L_BQ], 0, nullptr, 0, qb, DD, 0, NTOK, DD, DD, nullptr, 1);
    ttm_gemm(stream, EPI_BIAS, false, xln, DD, 0, F[lb + L_WK], DD, 0,
             F[lb + L_BK], 0, nullptr, 0, kb, DD, 0, NTOK, DD, DD, nullptr, 1);
    ttm_gemm(stream, EPI_BIAS, false, xln, DD, 0, F[lb + L_WV], DD, 0,
             F[lb + L_BV], 0, nullptr, 0, vb, DD, 0, NTOK, DD, DD, nullptr, 1);
    ttm_attention_kernel<<<BB * HH, 256, 0, stream>>>(qb, kb, vb, ab);
    // out-proj + residual accumulate into fused
    ttm_gemm(stream, EPI_RES, false, ab, DD, 0, F[lb + L_WO], DD, 0,
             F[lb + L_BO], 0, nullptr, 0, fused, DD, 0, NTOK, DD, DD, nullptr,
             1);
    // pre-MoE LN
    ttm_layernorm_kernel<<<NTOK, 256, 0, stream>>>(fused, F[lb + L_N2S],
                                                   F[lb + L_N2B], xln, DD);
    // router
    ttm_gate_kernel<<<(NTOK * EE + 255) / 256, 256, 0, stream>>>(
        xln, F[lb + L_GATE], logits, NTOK);
    ttm_zero_kernel<<<1, 32, 0, stream>>>(auxw, 17);  // keep auxw[17] running
    ttm_router_kernel<<<(NTOK + 255) / 256, 256, 0, stream>>>(logits, combine,
                                                              auxw, NTOK);
    ttm_aux_layer_finalize<<<1, 1, 0, stream>>>(auxw, (float)NTOK);
    // dense all-expert FFN, combine-weighted accumulate into fused
    for (int e = 0; e < EE; ++e) {
      ttm_gemm(stream, EPI_GELU, false, xln, DD, 0,
               F[lb + L_EW1] + (size_t)e * DD * FF_, FF_, 0,
               F[lb + L_EB1] + (size_t)e * FF_, 0, nullptr, 0, bufA, FF_, 0,
               NTOK, FF_, DD, nullptr, 1);
      ttm_gemm(stream, EPI_MOE, false, bufA, FF_, 0,
               F[lb + L_EW2] + (size_t)e * FF_ * DD, DD, 0,
               F[lb + L_EB2] + (size_t)e * DD, 0, combine + e, EE, fused, DD, 0,
               NTOK, DD, FF_, nullptr, 1);
    }
  }

  // ---- Head ------------------------------------------------------------
  ttm_layernorm_kernel<<<NTOK, 256, 0, stream>>>(fused, F[IDX_FN_S],
                                                 F[IDX_FN_B], xln, DD);
  ttm_pool_kernel<<<(NPTOK * DD + 255) / 256, 256, 0, stream>>>(xln, out_pool);
  // bottleneck: [B*T, D] @ [D, LR]
  ttm_gemm(stream, EPI_BIAS, false, out_pool, DD, 0, F[IDX_LR_W], LR_, 0,
           nullptr, 0, nullptr, 0, bottl, LR_, 0, NPTOK, LR_, DD, nullptr, 1);
  // subject readout: per-batch gathered weights, transposed store ->
  // prediction[b, v, t]
  ttm_gemm(stream, EPI_BIAS, true, bottl, LR_, (long)TT * LR_, F[IDX_SUBJ_W],
           NV_, (long)LR_ * NV_, F[IDX_SUBJ_B], (long)NV_, nullptr, 0, out_pred,
           TT, (long)NV_ * TT, TT, NV_, LR_, subject_id, BB);
  ttm_aux_write<<<1, 1, 0, stream>>>(auxw, out_aux);
}
